// _FCtL_36747740184603
// MI455X (gfx1250) — matrix operand order repaired and checked by reading, not compiled, not run on hardware
//
#include <hip/hip_runtime.h>
#include <hip/hip_bf16.h>

// ---------------------------------------------------------------------------
// FCtL attention block, fused flash-attention style for MI455X (gfx1250).
// B=1, C=256, H=W=96 -> N=M=9216 positions, head dim P=64, value dim 256.
// Heavy math on v_wmma_f32_16x16x32_bf16; K/V blocks staged into LDS by the
// Tensor Data Mover (tensor_load_to_lds), double-buffered, s_wait_tensorcnt.
// ---------------------------------------------------------------------------

typedef __bf16 bf16_t;
typedef __attribute__((ext_vector_type(16))) __bf16 v16bf;
typedef __attribute__((ext_vector_type(8)))  __bf16 v8bf;
typedef __attribute__((ext_vector_type(8)))  float  v8f;
typedef __attribute__((ext_vector_type(4)))  unsigned int v4u;
typedef __attribute__((ext_vector_type(8)))  int v8i;
typedef __attribute__((ext_vector_type(4)))  int v4i;

#define C_IN   256
#define N_POS  9216     // H*W
#define DH     64       // P (query/key dim)
#define NB     32       // query rows per workgroup
#define MB     64       // key/value block per iteration

// ---- WMMA helpers (contiguous k = 16*half + e for A and B, wave32) -------------------

static __device__ inline v8f wmma_bf16(v16bf a, v16bf b, v8f c) {
  // (neg_a, A, neg_b, B, c_mod, C, reuse_a, reuse_b)
  return __builtin_amdgcn_wmma_f32_16x16x32_bf16(false, a, false, b,
                                                 (short)0, c, false, false);
}

// A-matrix 16x32 bf16: lane holds row M=lane&15; K = half*16 + e
// (half = lane>>4). Source is row-major bf16 with leading dim ld.
static __device__ inline v16bf load_a_bf16(const bf16_t* base, int ld,
                                           int row0, int k0, int lane) {
  int row  = row0 + (lane & 15);
  int half = lane >> 4;
  const bf16_t* rp = base + (long)row * ld + k0 + half * 16;
  v8bf lo = *(const v8bf*)(rp);        // K = k0 + half*16 + 0..7
  v8bf hi = *(const v8bf*)(rp + 8);    // K = k0 + half*16 + 8..15
  return __builtin_shufflevector(lo, hi, 0,1,2,3,4,5,6,7,8,9,10,11,12,13,14,15);
}

// B-matrix 32x16 bf16: lane holds col N=lane&15; K = (lane>>4)*16 + e.
// base must already point at (K row 0, col 0) of the tile; ld = col stride.
static __device__ inline v16bf load_b_bf16(const bf16_t* base, long ld,
                                           int k0, int lane) {
  const bf16_t* p = base + (long)(lane & 15) * ld + k0 + (lane >> 4) * 16;
  return *(const v16bf*)p;
}

// ---- Tensor Data Mover: 2D tile (bf16 elements) -> LDS ---------------------
// D# per CDNA5 ISA 8.3-8.5. group0: count=1 | lds_addr | global_addr | type=2.
// group1: data_size=2B, tensor_dim0/1, tile_dim0/1, tensor_dim0_stride.
// Groups 2/3 (and trailing group) zero: <=2D tensor. 6-arg builtin, cpol=0.
static __device__ inline void tdm_load_2d(const void* gptr, unsigned lds_off,
                                          unsigned tdim0, unsigned tdim1,
                                          unsigned tile0, unsigned tile1,
                                          unsigned long long stride0) {
  unsigned long long ga = (unsigned long long)gptr;
  v4u g0 = { 1u,                                   // count=1 (valid, user mode)
             lds_off,                              // lds_addr (bytes)
             (unsigned)(ga & 0xffffffffull),       // global_addr[31:0]
             (unsigned)(((ga >> 32) & 0x01ffffffull) | 0x80000000u) }; // type=2
  v8i g1;
  g1[0] = 0x00010000;                              // data_size = 1 -> 2 bytes
  g1[1] = (int)((tdim0 & 0xffffu) << 16);          // tensor_dim0[15:0]
  g1[2] = (int)((tdim0 >> 16) | ((tdim1 & 0xffffu) << 16));
  g1[3] = (int)((tdim1 >> 16) | (tile0 << 16));    // tile_dim0
  g1[4] = (int)(tile1 & 0xffffu);                  // tile_dim1 (tile_dim2=0)
  g1[5] = (int)(stride0 & 0xffffffffull);          // tensor_dim0_stride[31:0]
  g1[6] = (int)((stride0 >> 32) & 0xffffull);      // stride[47:32] | d1stride=0
  g1[7] = 0;
  v4i z4 = { 0, 0, 0, 0 };
  v8i z8 = { 0, 0, 0, 0, 0, 0, 0, 0 };
  __builtin_amdgcn_tensor_load_to_lds(g0, g1, z4, z4, z8, 0);
}

// ---- kernel 1: channel means over spatial dim ------------------------------
__global__ void fctl_means_kernel(const float* __restrict__ x,
                                  const float* __restrict__ y,
                                  float* __restrict__ xmean,
                                  float* __restrict__ ymean) {
  __shared__ float red[256];
  int c = blockIdx.x & 255;
  const float* src = (blockIdx.x < 256) ? x : y;
  float*       dst = (blockIdx.x < 256) ? xmean : ymean;
  const float* row = src + (long)c * N_POS;
  float s = 0.f;
  for (int i = threadIdx.x; i < N_POS; i += 256) s += row[i];
  red[threadIdx.x] = s;
  __syncthreads();
  for (int off = 128; off > 0; off >>= 1) {
    if ((int)threadIdx.x < off) red[threadIdx.x] += red[threadIdx.x + off];
    __syncthreads();
  }
  if (threadIdx.x == 0) dst[c] = red[0] * (1.0f / N_POS);
}

// ---- kernel 2: Q/K projection, output [pos, 64] bf16 (centered) ------------
// out[n][p] = sum_c W[p][c] * (in[c][n] - mean[c]); bias cancels in centering.
__global__ void fctl_proj_pos_kernel(const float* __restrict__ in,
                                     const float* __restrict__ mean,
                                     const float* __restrict__ W,
                                     bf16_t* __restrict__ out) {
  int lane = threadIdx.x & 31;
  int wid  = threadIdx.x >> 5;
  int t    = blockIdx.x * 8 + wid;       // 0..2303
  int nt   = t >> 2;                     // 576 position tiles
  int pt   = t & 3;                      // 4 head-dim tiles
  int n0 = nt * 16, p0 = pt * 16;

  v8f acc = {};
  for (int kc = 0; kc < 8; ++kc) {
    int k0 = kc * 32;
    v16bf a;   // A = (x^T - mean) tile [16n x 32c]
    {
      int nrow = n0 + (lane & 15);
      int half = lane >> 4;
#pragma unroll
      for (int e = 0; e < 16; ++e) {
        int kk = half * 16 + e;
        int cc = k0 + kk;
        a[e] = (bf16_t)(in[(long)cc * N_POS + nrow] - mean[cc]);
      }
    }
    v16bf b;   // B = W^T tile [32c x 16p]: element(c,p) = W[p*256 + c]
    {
      const float* wp = W + (long)(p0 + (lane & 15)) * C_IN + k0 + (lane >> 4) * 16;
#pragma unroll
      for (int e = 0; e < 16; ++e) b[e] = (bf16_t)wp[e];
    }
    acc = wmma_bf16(a, b, acc);
  }
  int half = lane >> 4;
  int p = p0 + (lane & 15);
#pragma unroll
  for (int r = 0; r < 8; ++r) {
    int n = n0 + r + 8 * half;
    out[(long)n * DH + p] = (bf16_t)acc[r];
  }
}

// ---- kernel 3: V projection, output [256, M] bf16 --------------------------
__global__ void fctl_proj_cm_kernel(const float* __restrict__ y,
                                    const float* __restrict__ Wv,
                                    bf16_t* __restrict__ Vb) {
  int lane = threadIdx.x & 31;
  int wid  = threadIdx.x >> 5;
  int t    = blockIdx.x * 8 + wid;       // 0..9215
  int cot  = t / 576;
  int mt   = t % 576;
  int c0 = cot * 16, m0 = mt * 16;

  v8f acc = {};
  for (int kc = 0; kc < 8; ++kc) {
    int k0 = kc * 32;
    v16bf a;   // A = Wv tile [16co x 32c]
    {
      const float* wp = Wv + (long)(c0 + (lane & 15)) * C_IN + k0;
      int half = lane >> 4;
#pragma unroll
      for (int e = 0; e < 16; ++e) {
        int kk = half * 16 + e;
        a[e] = (bf16_t)wp[kk];
      }
    }
    v16bf b;   // B = y tile [32c x 16m]
    {
      int m = m0 + (lane & 15);
      int cbase = k0 + (lane >> 4) * 16;
#pragma unroll
      for (int e = 0; e < 16; ++e) b[e] = (bf16_t)y[(long)(cbase + e) * N_POS + m];
    }
    acc = wmma_bf16(a, b, acc);
  }
  int half = lane >> 4;
  int m = m0 + (lane & 15);
#pragma unroll
  for (int r = 0; r < 8; ++r) {
    int co = c0 + r + 8 * half;
    Vb[(long)co * N_POS + m] = (bf16_t)acc[r];
  }
}

// ---- kernel 4: fused flash attention with TDM double buffering -------------
// 288 WGs x 8 waves. Wave w: n-tile (w&1); S-phase m-tile (w>>1); PV-phase
// c-tiles (w>>1)*64..+63. K/V blocks DMA'd into LDS by the TDM; wave 0 owns
// the descriptors and TENSORcnt, workgroup barriers publish the data.
__global__ void fctl_attn_kernel(const bf16_t* __restrict__ Qb,
                                 const bf16_t* __restrict__ Kb,
                                 const bf16_t* __restrict__ Vb,
                                 const float* __restrict__ gamma,
                                 float* __restrict__ out) {
  __shared__ float  sS[NB][MB];
  __shared__ alignas(32) bf16_t sP[NB][MB];
  __shared__ alignas(32) bf16_t sK[2][MB * DH];     // [m_local][p], 2 x 8 KB
  __shared__ alignas(32) bf16_t sV[2][C_IN * MB];   // [c][m_local], 2 x 32 KB
  __shared__ float  sMax[NB], sSum[NB], sScale[NB];

  int lane  = threadIdx.x & 31;
  int wid   = threadIdx.x >> 5;
  int n_blk = blockIdx.x * NB;

  int nt   = wid & 1;    // accumulator / S-phase n-tile
  int mt   = wid >> 1;   // S-phase m-tile (0..3)
  int cgrp = wid >> 1;   // PV-phase channel group (4 c-tiles each)

  // Q fragments for this wave's 16 query rows (reused across all 144 blocks)
  v16bf qa0 = load_a_bf16(Qb, DH, n_blk + nt * 16, 0,  lane);
  v16bf qa1 = load_a_bf16(Qb, DH, n_blk + nt * 16, 32, lane);

  v8f zero = {};
  v8f acc[4];
#pragma unroll
  for (int j = 0; j < 4; ++j) acc[j] = zero;

  if (threadIdx.x < NB) { sMax[threadIdx.x] = -1e30f; sSum[threadIdx.x] = 0.f; }

  // TDM preload of block 0 (wave 0 issues; TENSORcnt is per-wave)
  if (wid == 0) {
    // K block: [M,64] layout makes a 64-row block fully contiguous (1D, 8 KB)
    tdm_load_2d(Kb, (unsigned)(uintptr_t)&sK[0][0],
                MB * DH, 1, MB * DH, 1, MB * DH);
    // V block: 2D tile, 256 rows x 64 bf16, row stride N_POS elements (32 KB)
    tdm_load_2d(Vb, (unsigned)(uintptr_t)&sV[0][0],
                N_POS, C_IN, MB, C_IN, N_POS);
  }
  __syncthreads();

  int it = 0;
  for (int m0 = 0; m0 < N_POS; m0 += MB, ++it) {
    int cur = it & 1;
    if (wid == 0) __builtin_amdgcn_s_wait_tensorcnt(0);
    __syncthreads();   // LDS block `cur` valid for all waves

    // prefetch next block into the other buffer (overlaps with WMMA below)
    if (wid == 0 && (m0 + MB) < N_POS) {
      tdm_load_2d(Kb + (long)(m0 + MB) * DH, (unsigned)(uintptr_t)&sK[cur ^ 1][0],
                  MB * DH, 1, MB * DH, 1, MB * DH);
      tdm_load_2d(Vb + (m0 + MB), (unsigned)(uintptr_t)&sV[cur ^ 1][0],
                  N_POS, C_IN, MB, C_IN, N_POS);
    }

    // ---- S tile = Q^T K (each wave: one 16x16 tile, K over P=64) ----
    {
      const bf16_t* kbase = &sK[cur][(mt * 16) * DH];
      v16bf kb0 = load_b_bf16(kbase, DH, 0,  lane);
      v16bf kb1 = load_b_bf16(kbase, DH, 32, lane);
      v8f s = zero;
      s = wmma_bf16(qa0, kb0, s);
      s = wmma_bf16(qa1, kb1, s);
      int half = lane >> 4;
      int ml = mt * 16 + (lane & 15);
#pragma unroll
      for (int r = 0; r < 8; ++r) sS[nt * 16 + r + 8 * half][ml] = s[r];
    }
    __syncthreads();

    // ---- online softmax: one lane per query row ----
    if (threadIdx.x < NB) {
      int row = threadIdx.x;
      float bm = -1e30f;
#pragma unroll 8
      for (int j = 0; j < MB; ++j) bm = fmaxf(bm, sS[row][j]);
      float om   = sMax[row];
      float nm   = fmaxf(om, bm);
      float corr = __expf(om - nm);
      float lsum = 0.f;
#pragma unroll 8
      for (int j = 0; j < MB; ++j) {
        float p = __expf(sS[row][j] - nm);
        sP[row][j] = (bf16_t)p;
        lsum += p;
      }
      sSum[row]   = sSum[row] * corr + lsum;
      sMax[row]   = nm;
      sScale[row] = corr;
    }
    __syncthreads();

    // ---- rescale accumulators, then acc += P x V (V from LDS) ----
    {
      int half = lane >> 4;
      float sc[8];
#pragma unroll
      for (int r = 0; r < 8; ++r) sc[r] = sScale[nt * 16 + r + 8 * half];
#pragma unroll
      for (int j = 0; j < 4; ++j)
#pragma unroll
        for (int r = 0; r < 8; ++r) acc[j][r] *= sc[r];

      v16bf pa0 = load_a_bf16(&sP[0][0], MB, nt * 16, 0,  lane);
      v16bf pa1 = load_a_bf16(&sP[0][0], MB, nt * 16, 32, lane);
#pragma unroll
      for (int j = 0; j < 4; ++j) {
        const bf16_t* vbase = &sV[cur][(cgrp * 64 + j * 16) * MB];
        v16bf vb0 = load_b_bf16(vbase, MB, 0,  lane);
        v16bf vb1 = load_b_bf16(vbase, MB, 32, lane);
        acc[j] = wmma_bf16(pa0, vb0, acc[j]);
        acc[j] = wmma_bf16(pa1, vb1, acc[j]);
      }
    }
    __syncthreads();
  }

  // ---- epilogue: out[c][n] = gamma * acc / sum ----
  float g = gamma[0];
  int half = lane >> 4;
#pragma unroll
  for (int j = 0; j < 4; ++j) {
    int c = cgrp * 64 + j * 16 + (lane & 15);
#pragma unroll
    for (int r = 0; r < 8; ++r) {
      int nl = nt * 16 + r + 8 * half;
      float inv = 1.0f / sSum[nl];
      out[(long)c * N_POS + (n_blk + nl)] = g * acc[j][r] * inv;
    }
  }
}

// ---------------------------------------------------------------------------

extern "C" void kernel_launch(void* const* d_in, const int* in_sizes, int n_in,
                              void* d_out, int out_size, void* d_ws, size_t ws_size,
                              hipStream_t stream) {
  // setup_inputs order: x, y, Wv, Wq, bq, Wk, bk, gamma
  const float* x     = (const float*)d_in[0];
  const float* y     = (const float*)d_in[1];
  const float* Wv    = (const float*)d_in[2];
  const float* Wq    = (const float*)d_in[3];
  const float* Wk    = (const float*)d_in[5];
  const float* gamma = (const float*)d_in[7];
  // bq/bk are mathematically cancelled by mean-centering; unused.

  char* ws = (char*)d_ws;
  bf16_t* Qb    = (bf16_t*)(ws);                 // [N,64]  bf16  1,179,648 B
  bf16_t* Kb    = (bf16_t*)(ws + 1179648);       // [M,64]  bf16  1,179,648 B
  bf16_t* Vb    = (bf16_t*)(ws + 2359296);       // [256,M] bf16  4,718,592 B
  float*  xmean = (float*)(ws + 7077888);        // 256 f32
  float*  ymean = (float*)(ws + 7078912);        // 256 f32  (total ~7.08 MB)

  fctl_means_kernel<<<512, 256, 0, stream>>>(x, y, xmean, ymean);
  fctl_proj_pos_kernel<<<288, 256, 0, stream>>>(x, xmean, Wq, Qb);   // Q (centered)
  fctl_proj_pos_kernel<<<288, 256, 0, stream>>>(y, ymean, Wk, Kb);   // K (centered)
  fctl_proj_cm_kernel<<<1152, 256, 0, stream>>>(y, Wv, Vb);          // V
  fctl_attn_kernel<<<288, 256, 0, stream>>>(Qb, Kb, Vb, gamma, (float*)d_out);
}
